// OutputHeads_77464030151266
// MI455X (gfx1250) — matrix operand order repaired and checked by reading, not compiled, not run on hardware
//
#include <hip/hip_runtime.h>
#include <math.h>

// Problem constants (from reference)
#define DIM    2048
#define VOCAB  50257
#define MTOT   4096          // BATCH*SEQ = 2*2048

// Tiling
#define BM 128
#define BN 128
#define BK 32
#define SSTR 40              // LDS row stride in f16 elems (BK + 8 pad, keeps 16B align)
#define NBLK_RED 1024

typedef __attribute__((ext_vector_type(16))) _Float16 v16h;
typedef __attribute__((ext_vector_type(8)))  _Float16 v8h;
typedef __attribute__((ext_vector_type(4)))  _Float16 v4h;
typedef __attribute__((ext_vector_type(8)))  float    v8f;
typedef __attribute__((ext_vector_type(4)))  float    v4f;

// ---------------------------------------------------------------------------
// Pass 1a: per-block partial sums of |w|  (deterministic: fixed block slots)
// ---------------------------------------------------------------------------
__global__ void __launch_bounds__(256)
absmean_partial_k(const float* __restrict__ w, float* __restrict__ partial, int n4)
{
    __shared__ float red[256];
    const v4f* w4 = (const v4f*)w;
    float s = 0.0f;
    for (int i = blockIdx.x * 256 + threadIdx.x; i < n4; i += gridDim.x * 256) {
        v4f v = w4[i];
        s += fabsf(v[0]) + fabsf(v[1]) + fabsf(v[2]) + fabsf(v[3]);
    }
    red[threadIdx.x] = s;
    __syncthreads();
    for (int off = 128; off > 0; off >>= 1) {
        if (threadIdx.x < off) red[threadIdx.x] += red[threadIdx.x + off];
        __syncthreads();
    }
    if (threadIdx.x == 0) partial[blockIdx.x] = red[0];
}

// ---------------------------------------------------------------------------
// Pass 1b: final deterministic reduction -> gamma = max(mean|w|, 1e-5), 1/gamma
// ---------------------------------------------------------------------------
__global__ void __launch_bounds__(256)
absmean_final_k(const float* __restrict__ partial, float* __restrict__ gbuf, float inv_count)
{
    __shared__ float red[256];
    float s = 0.0f;
    for (int i = threadIdx.x; i < NBLK_RED; i += 256) s += partial[i];
    red[threadIdx.x] = s;
    __syncthreads();
    for (int off = 128; off > 0; off >>= 1) {
        if (threadIdx.x < off) red[threadIdx.x] += red[threadIdx.x + off];
        __syncthreads();
    }
    if (threadIdx.x == 0) {
        float gamma = fmaxf(red[0] * inv_count, 1e-5f);
        gbuf[0] = gamma;
        gbuf[1] = 1.0f / gamma;
    }
}

// ---------------------------------------------------------------------------
// Pass 2: out = gamma * (x_f16 @ ternary(w)^T), WMMA f32_16x16x32_f16
//   block = 128(M) x 128(N), 256 threads = 8 waves as 2(M) x 4(N)
//   each wave: 64x32 = 4x2 tiles of 16x16
// ---------------------------------------------------------------------------
__global__ void __launch_bounds__(256)
bitlinear_wmma_k(const float* __restrict__ x, const float* __restrict__ w,
                 const float* __restrict__ gbuf, float* __restrict__ out)
{
    __shared__ _Float16 sA[BM * SSTR];   // x tile, f16
    __shared__ _Float16 sB[BN * SSTR];   // ternarized w tile, f16

    const float gamma     = gbuf[0];
    const float inv_gamma = gbuf[1];

    const int mBase = blockIdx.x * BM;   // m-fastest linearization: consecutive
    const int nBase = blockIdx.y * BN;   // blocks share the same w n-tile (L2)

    const int tid   = threadIdx.x;
    const int lane  = tid & 31;
    const int wid   = tid >> 5;          // 0..7
    const int waveM = wid >> 2;          // 0..1
    const int waveN = wid & 3;           // 0..3
    const int l16   = lane & 15;
    const int khalf = lane >> 4;         // 0 or 1

    v8f acc[4][2] = {};

    for (int k0 = 0; k0 < DIM; k0 += BK) {
        __syncthreads();

        // ---- cooperative stage: x tile (fp32 -> f16) ----
        #pragma unroll
        for (int it = 0; it < 4; ++it) {
            int f   = tid + it * 256;        // 0..1023 float4 slots
            int row = f >> 3;                // 0..127
            int c4  = (f & 7) << 2;          // 0,4,...,28
            v4f a = *(const v4f*)(x + (size_t)(mBase + row) * DIM + k0 + c4);
            v4h h = { (_Float16)a[0], (_Float16)a[1], (_Float16)a[2], (_Float16)a[3] };
            *(v4h*)(&sA[row * SSTR + c4]) = h;
        }
        // ---- cooperative stage: w tile (fp32 -> ternary f16), n-guarded ----
        #pragma unroll
        for (int it = 0; it < 4; ++it) {
            int f   = tid + it * 256;
            int row = f >> 3;
            int c4  = (f & 7) << 2;
            int v   = nBase + row;
            v4f b = { 0.0f, 0.0f, 0.0f, 0.0f };
            if (v < VOCAB) {
                b = *(const v4f*)(w + (size_t)v * DIM + k0 + c4);
                if (k0 + BK < DIM)  // stream-ahead hint for next K-step
                    __builtin_prefetch(w + (size_t)v * DIM + k0 + BK + c4, 0, 1);
            }
            v4h h;
            #pragma unroll
            for (int e = 0; e < 4; ++e) {
                float t = __builtin_rintf(b[e] * inv_gamma);   // round-half-even, like jnp.round
                t = fminf(1.0f, fmaxf(-1.0f, t));
                h[e] = (_Float16)t;
            }
            *(v4h*)(&sB[row * SSTR + c4]) = h;
        }
        __syncthreads();

        // ---- fragments (ISA 7.12.2 layouts) ----
        union F16F { v16h v; v8h h[2]; };
        F16F afrag[4], bfrag[2];
        #pragma unroll
        for (int i = 0; i < 4; ++i) {
            // A 16x32 f16: lanes<16 K∈{0..7,16..23}, lanes>=16 K∈{8..15,24..31}
            int row = waveM * 64 + i * 16 + l16;
            afrag[i].h[0] = *(const v8h*)(&sA[row * SSTR + khalf * 8]);
            afrag[i].h[1] = *(const v8h*)(&sA[row * SSTR + 16 + khalf * 8]);
        }
        #pragma unroll
        for (int j = 0; j < 2; ++j) {
            // B 32x16 f16: lanes<16 K∈{0..7,16..23}, lanes>=16 K∈{8..15,24..31}
            int col = waveN * 32 + j * 16 + l16;
            bfrag[j].h[0] = *(const v8h*)(&sB[col * SSTR + khalf * 8]);
            bfrag[j].h[1] = *(const v8h*)(&sB[col * SSTR + 16 + khalf * 8]);
        }

        // ---- 8 WMMAs per K-step ----
        #pragma unroll
        for (int i = 0; i < 4; ++i)
            #pragma unroll
            for (int j = 0; j < 2; ++j)
                acc[i][j] = __builtin_amdgcn_wmma_f32_16x16x32_f16(
                    false, afrag[i].v, false, bfrag[j].v,
                    (short)0, acc[i][j], false, false);
    }

    // ---- epilogue: scale by gamma, store (C layout: N=l16, M=r+8*khalf) ----
    #pragma unroll
    for (int i = 0; i < 4; ++i) {
        #pragma unroll
        for (int j = 0; j < 2; ++j) {
            int n = nBase + waveN * 32 + j * 16 + l16;
            if (n < VOCAB) {
                int mrow = mBase + waveM * 64 + i * 16 + 8 * khalf;
                #pragma unroll
                for (int r = 0; r < 8; ++r) {
                    out[(size_t)(mrow + r) * VOCAB + n] = gamma * acc[i][j][r];
                }
            }
        }
    }
}

// ---------------------------------------------------------------------------
extern "C" void kernel_launch(void* const* d_in, const int* in_sizes, int n_in,
                              void* d_out, int out_size, void* d_ws, size_t ws_size,
                              hipStream_t stream)
{
    const float* x = (const float*)d_in[0];   // [4096, 2048] fp32
    const float* w = (const float*)d_in[1];   // [50257, 2048] fp32
    float* out = (float*)d_out;               // [4096, 50257] fp32
    float* ws  = (float*)d_ws;

    float* partial = ws;                      // NBLK_RED floats
    float* gbuf    = ws + NBLK_RED;           // gamma, inv_gamma

    const long n  = (long)VOCAB * DIM;        // 102,926,336 (divisible by 4)
    const int  n4 = (int)(n / 4);

    absmean_partial_k<<<NBLK_RED, 256, 0, stream>>>(w, partial, n4);
    absmean_final_k<<<1, 256, 0, stream>>>(partial, gbuf, 1.0f / (float)n);

    dim3 grid(MTOT / BM, (VOCAB + BN - 1) / BN);   // 32 x 393, m-fastest
    bitlinear_wmma_k<<<grid, 256, 0, stream>>>(x, w, gbuf, out);
}
